// PointNetFeaturePropagation_3143916061382
// MI455X (gfx1250) — matrix operand order repaired and checked by reading, not compiled, not run on hardware
//
#include <hip/hip_runtime.h>

#define B_   8
#define N_   8192
#define S_   2048
#define C1_  128
#define C2_  256
#define CIN_ 384
#define H_   256
#define P_   65536          // B_*N_
#define EPS_INTERP 1e-8f
#define EPS_BN     1e-5f

#define LDS_STRIDE 40       // halves per row (80 B): conflict-free b128 frag loads

typedef __attribute__((ext_vector_type(16))) __bf16 v16bf;
typedef __attribute__((ext_vector_type(8)))  float  v8f;
typedef __attribute__((ext_vector_type(4)))  unsigned int v4u_t;
typedef __attribute__((ext_vector_type(8)))  int v8i_t;
typedef __attribute__((ext_vector_type(4)))  int v4i_t;

typedef union { v4u_t q[2]; unsigned int u32[8]; unsigned short u16[16]; v16bf v; } BfFrag;

// --------------------------- TDM availability ------------------------------
// ROCm 7.2 (clang-22):      5-arg builtin (v4u, v8i, v4i, v4i, i32)
// amdgpu-toolchain (c23):   6-arg builtin (v4u, v8i, v4i, v4i, v8i, i32);
//                           identified by its gfx1250 TDM header shipping.
#if defined(__has_builtin)
#  if __has_builtin(__builtin_amdgcn_tensor_load_to_lds)
#    if __has_include(<hip/amd_detail/amd_gfx1250_TDM.h>)
#      define TDM_ARITY 6
#    else
#      define TDM_ARITY 5
#    endif
#  endif
#endif
#ifndef TDM_ARITY
#  define TDM_ARITY 0
#endif
#define USE_TDM (TDM_ARITY != 0)

#if USE_TDM
// 2-D bf16 tile load Global->LDS via Tensor Data Mover, with LDS row padding:
// rows of tile_k elements (tile_k*2 = 64 bytes here) padded to 80 B.
// D# layout per CDNA5 ISA §8.3-8.6.
__device__ __forceinline__ void tdm_load_tile_2d(unsigned lds_byte_off,
                                                 unsigned long long gaddr,
                                                 unsigned tile_k,     // dim0 elements (=32)
                                                 unsigned tile_m,     // dim1 rows   (=128)
                                                 unsigned stride_el,  // tensor_dim0_stride
                                                 unsigned tens_d0,
                                                 unsigned tens_d1) {
  v4u_t g0;
  g0[0] = 1u;                                         // count=1 (valid user D#)
  g0[1] = lds_byte_off;                               // lds_addr
  g0[2] = (unsigned)(gaddr & 0xFFFFFFFFu);            // global_addr[31:0]
  g0[3] = (unsigned)((gaddr >> 32) & 0x1FFFFFFu)      // global_addr[56:32]
        | (2u << 30);                                 // type=2 ("image")
  v8i_t g1;
  g1[0] = (int)((1u << 16)        // data_size = 1 -> 2 bytes
              | (1u << 20)        // pad_enable
              | (3u << 22)        // pad_interval: every 16 dwords (64 B)
              | (3u << 25));      // pad_amount:   4 dwords (16 B)
  g1[1] = (int)((tens_d0 & 0xFFFFu) << 16);                            // tensor_dim0 lo
  g1[2] = (int)(((tens_d0 >> 16) & 0xFFFFu) | ((tens_d1 & 0xFFFFu) << 16));
  g1[3] = (int)(((tens_d1 >> 16) & 0xFFFFu) | (tile_k << 16));         // tile_dim0
  g1[4] = (int)(tile_m & 0xFFFFu);                                     // tile_dim1 (dim2=0)
  g1[5] = (int)stride_el;                                              // dim0_stride[31:0]
  g1[6] = 0;
  g1[7] = 0;
  v4i_t gz; gz[0] = 0; gz[1] = 0; gz[2] = 0; gz[3] = 0;                // 2-D: groups 2/3 unused
#if TDM_ARITY == 5
  __builtin_amdgcn_tensor_load_to_lds(g0, g1, gz, gz, 0);
#else
  v8i_t gz8; gz8[0] = 0; gz8[1] = 0; gz8[2] = 0; gz8[3] = 0;
  gz8[4] = 0; gz8[5] = 0; gz8[6] = 0; gz8[7] = 0;
  __builtin_amdgcn_tensor_load_to_lds(g0, g1, gz, gz, gz8, 0);
#endif
}
#endif

__device__ __forceinline__ unsigned short f2bf(float x) {
  unsigned int u = __float_as_uint(x);
  u += 0x7FFFu + ((u >> 16) & 1u);          // round-to-nearest-even
  return (unsigned short)(u >> 16);
}

__device__ __forceinline__ v8f v8f_zero() {
  v8f z;
#pragma unroll
  for (int i = 0; i < 8; ++i) z[i] = 0.0f;
  return z;
}

// A fragment: 16x32 bf16 tile, LDS [m][k], row stride LDS_STRIDE halves.
// ISA 7.12.2: lane m=lane&15, kbase=(lane>=16)?8:0; VGPR0-3 = k kb..kb+7,
// VGPR4-7 = k kb+16..kb+23  -> two contiguous 16B runs -> 2x ds_load_b128.
__device__ __forceinline__ v16bf load_frag_a(const unsigned short* lds, int lane) {
  const int m = lane & 15;
  const int kb = (lane >> 4) << 3;
  BfFrag f;
  f.q[0] = *(const v4u_t*)(lds + m * LDS_STRIDE + kb);
  f.q[1] = *(const v4u_t*)(lds + m * LDS_STRIDE + kb + 16);
  return f.v;
}

// B fragment from TRANSPOSED tile: LDS [n][k], k contiguous.
// lane n=lane&15, kbase=(lane>=16)?8:0; k kb..kb+7 and kb+16..kb+23
// -> two contiguous 16B runs -> 2x ds_load_b128.
__device__ __forceinline__ v16bf load_frag_bT(const unsigned short* lds, int lane) {
  const int n = lane & 15;
  const int kb = (lane >> 4) << 3;
  BfFrag f;
  f.q[0] = *(const v4u_t*)(lds + n * LDS_STRIDE + kb);
  f.q[1] = *(const v4u_t*)(lds + n * LDS_STRIDE + kb + 16);
  return f.v;
}

// ---------------------------------------------------------------------------
// Kernel 1: weight conversion f32 -> bf16 (w1: 256x384, w2: 256x256)
// ---------------------------------------------------------------------------
__global__ void k_cvt_w(const float* __restrict__ w1, const float* __restrict__ w2,
                        unsigned short* __restrict__ W1b, unsigned short* __restrict__ W2b) {
  int i = blockIdx.x * 256 + threadIdx.x;
  if (i < H_ * CIN_) W1b[i] = f2bf(w1[i]);
  int j = i - H_ * CIN_;
  if (j >= 0 && j < H_ * H_) W2b[j] = f2bf(w2[j]);
}

// ---------------------------------------------------------------------------
// Kernel 2: 3-NN + inverse-distance weights; xyz2 batch staged in LDS (24 KB)
// ---------------------------------------------------------------------------
__global__ __launch_bounds__(256) void k_knn(const float* __restrict__ xyz1,
                                             const float* __restrict__ xyz2,
                                             int* __restrict__ idx, float* __restrict__ wgt) {
  __shared__ float s2x[S_], s2y[S_], s2z[S_];
  const int p = blockIdx.x * 256 + threadIdx.x;   // N_=8192 % 256 == 0 -> one batch/block
  const int b = p >> 13;
  for (int i = threadIdx.x; i < S_; i += 256) {
    const float* q = xyz2 + ((size_t)b * S_ + i) * 3;
    s2x[i] = q[0]; s2y[i] = q[1]; s2z[i] = q[2];
  }
  __syncthreads();

  const float px = xyz1[p * 3 + 0], py = xyz1[p * 3 + 1], pz = xyz1[p * 3 + 2];
  float d0 = 3.4e38f, d1 = 3.4e38f, d2 = 3.4e38f;
  int   i0 = 0, i1 = 0, i2 = 0;
  for (int s = 0; s < S_; ++s) {
    const float dx = px - s2x[s], dy = py - s2y[s], dz = pz - s2z[s];
    const float d = dx * dx + dy * dy + dz * dz;
    if (d < d2) {
      if (d < d1) {
        if (d < d0) { d2 = d1; i2 = i1; d1 = d0; i1 = i0; d0 = d; i0 = s; }
        else        { d2 = d1; i2 = i1; d1 = d;  i1 = s; }
      } else        { d2 = d;  i2 = s; }
    }
  }
  const float r0 = 1.0f / (d0 + EPS_INTERP);
  const float r1 = 1.0f / (d1 + EPS_INTERP);
  const float r2 = 1.0f / (d2 + EPS_INTERP);
  const float inv = 1.0f / (r0 + r1 + r2);
  idx[p * 3 + 0] = i0; idx[p * 3 + 1] = i1; idx[p * 3 + 2] = i2;
  wgt[p * 3 + 0] = r0 * inv; wgt[p * 3 + 1] = r1 * inv; wgt[p * 3 + 2] = r2 * inv;
}

// ---------------------------------------------------------------------------
// Kernel 3: build X bf16 [c][p]  (c<128: copy points1, c>=128: interpolate)
// ---------------------------------------------------------------------------
__global__ __launch_bounds__(256) void k_build_x(const float* __restrict__ p1,
                                                 const float* __restrict__ p2,
                                                 const int* __restrict__ idx,
                                                 const float* __restrict__ wgt,
                                                 unsigned short* __restrict__ X) {
  const int p = blockIdx.x * 256 + threadIdx.x;
  const int c = blockIdx.y;
  const int b = p >> 13, n = p & (N_ - 1);
  float v;
  if (c < C1_) {
    v = p1[((size_t)b * C1_ + c) * N_ + n];
  } else {
    const int c2 = c - C1_;
    const float* row = p2 + ((size_t)b * C2_ + c2) * S_;
    v = wgt[p * 3 + 0] * row[idx[p * 3 + 0]]
      + wgt[p * 3 + 1] * row[idx[p * 3 + 1]]
      + wgt[p * 3 + 2] * row[idx[p * 3 + 2]];
  }
  X[(size_t)c * P_ + p] = f2bf(v);
}

// ---------------------------------------------------------------------------
// Kernel 4: GEMM1  y1[h,p] = sum_c W1[h,c] * X[c,p]   (bf16 WMMA, f32 accum)
// 128x128 block tile, 8 waves, wave tile 32(M)x64(N), K step 32.
// A tile via TDM (padded rows); B tile transposed into LDS by 8 waves.
// Conv bias skipped: it cancels exactly in the following BatchNorm.
// ---------------------------------------------------------------------------
__global__ __launch_bounds__(256) void k_gemm1(const unsigned short* __restrict__ Wb, // [H][CIN]
                                               const unsigned short* __restrict__ X,  // [CIN][P]
                                               float* __restrict__ Y) {               // [H][P]
  __shared__ __align__(16) unsigned short lA[128 * LDS_STRIDE];  // [m][k]
  __shared__ __align__(16) unsigned short lBt[128 * LDS_STRIDE]; // [n][k] (transposed)
  const int tid = threadIdx.x, lane = tid & 31, wave = tid >> 5;
  const int wm = wave & 3, wn = wave >> 2;
  const int pTile = blockIdx.x * 128, hTile = blockIdx.y * 128;

  v8f acc[2][4];
#pragma unroll
  for (int i = 0; i < 2; ++i)
#pragma unroll
    for (int j = 0; j < 4; ++j) acc[i][j] = v8f_zero();

  for (int k0 = 0; k0 < CIN_; k0 += 32) {
#if USE_TDM
    if (wave == 0) {
      tdm_load_tile_2d((unsigned)(size_t)(void*)lA,
                       (unsigned long long)(size_t)(const void*)(Wb + (size_t)hTile * CIN_ + k0),
                       32u, 128u, (unsigned)CIN_, (unsigned)CIN_, (unsigned)H_);
      __builtin_amdgcn_s_wait_tensorcnt(0);
    }
#else
#pragma unroll
    for (int it = 0; it < 8; ++it) {            // A: 128x32 = 2048 dwords
      const int e = tid + it * 256, row = e >> 4, kp = e & 15;
      *(unsigned int*)(&lA[row * LDS_STRIDE + 2 * kp]) =
          *(const unsigned int*)(Wb + (hTile + row) * CIN_ + k0 + 2 * kp);
    }
#endif
#pragma unroll
    for (int it = 0; it < 8; ++it) {            // B: 32x128, transpose into [n][k]
      const int e = tid + it * 256, row = e >> 6, np = e & 63;
      const unsigned int val =
          *(const unsigned int*)(X + (size_t)(k0 + row) * P_ + pTile + 2 * np);
      lBt[(2 * np)     * LDS_STRIDE + row] = (unsigned short)(val & 0xFFFFu);
      lBt[(2 * np + 1) * LDS_STRIDE + row] = (unsigned short)(val >> 16);
    }
    __syncthreads();
    if (k0 + 32 < CIN_)                          // gfx1250 global_prefetch_b8
      __builtin_prefetch(X + (size_t)(k0 + 32 + (lane & 31)) * P_ + pTile, 0, 0);

    v16bf a[2], bfr[4];
#pragma unroll
    for (int i = 0; i < 2; ++i) a[i] = load_frag_a(lA + (wm * 32 + i * 16) * LDS_STRIDE, lane);
#pragma unroll
    for (int j = 0; j < 4; ++j) bfr[j] = load_frag_bT(lBt + (wn * 64 + j * 16) * LDS_STRIDE, lane);
#pragma unroll
    for (int i = 0; i < 2; ++i)
#pragma unroll
      for (int j = 0; j < 4; ++j)
        acc[i][j] = __builtin_amdgcn_wmma_f32_16x16x32_bf16(
            false, a[i], false, bfr[j], (short)0, acc[i][j], false, false);
    __syncthreads();
  }
#pragma unroll
  for (int i = 0; i < 2; ++i)
#pragma unroll
    for (int j = 0; j < 4; ++j) {
      const int h = hTile + wm * 32 + i * 16 + ((lane >> 4) << 3);
      const int pp = pTile + wn * 64 + j * 16 + (lane & 15);
#pragma unroll
      for (int r = 0; r < 8; ++r) Y[(size_t)(h + r) * P_ + pp] = acc[i][j][r];
    }
}

// ---------------------------------------------------------------------------
// Kernel 5: per-channel sum / sumsq, deterministic (one block per channel).
// layout 0: Y[h*P + p]   layout 1: Y[b*H*N + h*N + n]
// ---------------------------------------------------------------------------
__global__ __launch_bounds__(256) void k_bn_stats(const float* __restrict__ Y,
                                                  float* __restrict__ sum, float* __restrict__ sq,
                                                  int layout) {
  __shared__ float ss[256], sz[256];
  const int h = blockIdx.x;
  float ls = 0.0f, lq = 0.0f;
  for (int p = threadIdx.x; p < P_; p += 256) {
    const size_t a = layout
        ? ((size_t)(p >> 13) * (H_ * N_) + (size_t)h * N_ + (p & (N_ - 1)))
        : ((size_t)h * P_ + p);
    const float v = Y[a];
    ls += v; lq += v * v;
  }
  ss[threadIdx.x] = ls; sz[threadIdx.x] = lq;
  __syncthreads();
  for (int s = 128; s > 0; s >>= 1) {
    if (threadIdx.x < s) { ss[threadIdx.x] += ss[threadIdx.x + s]; sz[threadIdx.x] += sz[threadIdx.x + s]; }
    __syncthreads();
  }
  if (threadIdx.x == 0) { sum[h] = ss[0]; sq[h] = sz[0]; }
}

__global__ void k_bn_finalize(const float* __restrict__ sum, const float* __restrict__ sq,
                              const float* __restrict__ g, const float* __restrict__ be,
                              float* __restrict__ scale, float* __restrict__ shift) {
  const int h = threadIdx.x;
  const float mean = sum[h] * (1.0f / P_);
  const float var  = sq[h] * (1.0f / P_) - mean * mean;
  const float sc   = g[h] * rsqrtf(var + EPS_BN);
  scale[h] = sc;
  shift[h] = be[h] - mean * sc;
}

// ---------------------------------------------------------------------------
// Kernel 6: GEMM2 with BN1+ReLU+bf16 fused into the B-tile transpose staging.
// Writes raw y2 into d_out laid out (B,H,N); BN2+ReLU applied in-place later.
// ---------------------------------------------------------------------------
__global__ __launch_bounds__(256) void k_gemm2(const unsigned short* __restrict__ Wb, // [H][H]
                                               const float* __restrict__ Y1,          // [H][P]
                                               const float* __restrict__ scale1,
                                               const float* __restrict__ shift1,
                                               float* __restrict__ Out) {             // (B,H,N)
  __shared__ __align__(16) unsigned short lA[128 * LDS_STRIDE];
  __shared__ __align__(16) unsigned short lBt[128 * LDS_STRIDE];
  __shared__ float sScale[H_], sShift[H_];
  const int tid = threadIdx.x, lane = tid & 31, wave = tid >> 5;
  const int wm = wave & 3, wn = wave >> 2;
  const int pTile = blockIdx.x * 128, hTile = blockIdx.y * 128;

  sScale[tid] = scale1[tid]; sShift[tid] = shift1[tid];   // H_ == 256 == blockDim
  __syncthreads();

  v8f acc[2][4];
#pragma unroll
  for (int i = 0; i < 2; ++i)
#pragma unroll
    for (int j = 0; j < 4; ++j) acc[i][j] = v8f_zero();

  for (int k0 = 0; k0 < H_; k0 += 32) {
#if USE_TDM
    if (wave == 0) {
      tdm_load_tile_2d((unsigned)(size_t)(void*)lA,
                       (unsigned long long)(size_t)(const void*)(Wb + (size_t)hTile * H_ + k0),
                       32u, 128u, (unsigned)H_, (unsigned)H_, (unsigned)H_);
      __builtin_amdgcn_s_wait_tensorcnt(0);
    }
#else
#pragma unroll
    for (int it = 0; it < 8; ++it) {
      const int e = tid + it * 256, row = e >> 4, kp = e & 15;
      *(unsigned int*)(&lA[row * LDS_STRIDE + 2 * kp]) =
          *(const unsigned int*)(Wb + (hTile + row) * H_ + k0 + 2 * kp);
    }
#endif
#pragma unroll
    for (int it = 0; it < 16; ++it) {           // B: 32x128 f32 -> BN -> ReLU -> bf16, transposed
      const int e = tid + it * 256, row = e >> 7, col = e & 127;
      const int k = k0 + row;
      float v = Y1[(size_t)k * P_ + pTile + col];
      v = v * sScale[k] + sShift[k];
      v = v > 0.0f ? v : 0.0f;
      lBt[col * LDS_STRIDE + row] = f2bf(v);
    }
    __syncthreads();
    if (k0 + 32 < H_)
      __builtin_prefetch(Y1 + (size_t)(k0 + 32 + (lane & 31)) * P_ + pTile, 0, 0);

    v16bf a[2], bfr[4];
#pragma unroll
    for (int i = 0; i < 2; ++i) a[i] = load_frag_a(lA + (wm * 32 + i * 16) * LDS_STRIDE, lane);
#pragma unroll
    for (int j = 0; j < 4; ++j) bfr[j] = load_frag_bT(lBt + (wn * 64 + j * 16) * LDS_STRIDE, lane);
#pragma unroll
    for (int i = 0; i < 2; ++i)
#pragma unroll
      for (int j = 0; j < 4; ++j)
        acc[i][j] = __builtin_amdgcn_wmma_f32_16x16x32_bf16(
            false, a[i], false, bfr[j], (short)0, acc[i][j], false, false);
    __syncthreads();
  }
#pragma unroll
  for (int i = 0; i < 2; ++i)
#pragma unroll
    for (int j = 0; j < 4; ++j) {
      const int h = hTile + wm * 32 + i * 16 + ((lane >> 4) << 3);
      const int pp = pTile + wn * 64 + j * 16 + (lane & 15);
      const int bb = pp >> 13, nn = pp & (N_ - 1);
#pragma unroll
      for (int r = 0; r < 8; ++r)
        Out[(size_t)bb * (H_ * N_) + (size_t)(h + r) * N_ + nn] = acc[i][j][r];
    }
}

// ---------------------------------------------------------------------------
// Kernel 7: in-place BN2 + ReLU on d_out (B,H,N)
// ---------------------------------------------------------------------------
__global__ void k_bn_apply(float* __restrict__ Y, const float* __restrict__ scale,
                           const float* __restrict__ shift) {
  const int i = blockIdx.x * 256 + threadIdx.x;      // 16,777,216 elements
  const int h = (i >> 13) & (H_ - 1);
  float v = Y[i] * scale[h] + shift[h];
  Y[i] = v > 0.0f ? v : 0.0f;
}

// ---------------------------------------------------------------------------
extern "C" void kernel_launch(void* const* d_in, const int* in_sizes, int n_in,
                              void* d_out, int out_size, void* d_ws, size_t ws_size,
                              hipStream_t stream) {
  const float* xyz1 = (const float*)d_in[0];
  const float* xyz2 = (const float*)d_in[1];
  const float* pts1 = (const float*)d_in[2];
  const float* pts2 = (const float*)d_in[3];
  const float* w1   = (const float*)d_in[4];
  // d_in[5] = b1, d_in[9] = b2: conv bias cancels exactly under BatchNorm
  const float* g1   = (const float*)d_in[6];
  const float* be1  = (const float*)d_in[7];
  const float* w2   = (const float*)d_in[8];
  const float* g2   = (const float*)d_in[10];
  const float* be2  = (const float*)d_in[11];
  float* out = (float*)d_out;

  char* ws = (char*)d_ws;
  size_t off = 0;
  auto take = [&](size_t bytes) -> void* {
    void* p = (void*)(ws + off);
    off = (off + bytes + 255) & ~(size_t)255;
    return p;
  };
  int*            idx  = (int*)take((size_t)P_ * 3 * sizeof(int));
  float*          wgt  = (float*)take((size_t)P_ * 3 * sizeof(float));
  unsigned short* W1b  = (unsigned short*)take((size_t)H_ * CIN_ * 2);
  unsigned short* W2b  = (unsigned short*)take((size_t)H_ * H_ * 2);
  unsigned short* Xb   = (unsigned short*)take((size_t)CIN_ * P_ * 2);
  float*          y1   = (float*)take((size_t)H_ * P_ * sizeof(float));
  float*          st   = (float*)take(2048 * sizeof(float));
  float *sum1 = st, *sq1 = st + 256, *scale1 = st + 512, *shift1 = st + 768;
  float *sum2 = st + 1024, *sq2 = st + 1280, *scale2 = st + 1536, *shift2 = st + 1792;

  k_cvt_w<<<(H_ * CIN_ + H_ * H_) / 256, 256, 0, stream>>>(w1, w2, W1b, W2b);
  k_knn<<<P_ / 256, 256, 0, stream>>>(xyz1, xyz2, idx, wgt);
  k_build_x<<<dim3(P_ / 256, CIN_), 256, 0, stream>>>(pts1, pts2, idx, wgt, Xb);
  k_gemm1<<<dim3(P_ / 128, H_ / 128), 256, 0, stream>>>(W1b, Xb, y1);
  k_bn_stats<<<H_, 256, 0, stream>>>(y1, sum1, sq1, 0);
  k_bn_finalize<<<1, H_, 0, stream>>>(sum1, sq1, g1, be1, scale1, shift1);
  k_gemm2<<<dim3(P_ / 128, H_ / 128), 256, 0, stream>>>(W2b, y1, scale1, shift1, out);
  k_bn_stats<<<H_, 256, 0, stream>>>(out, sum2, sq2, 1);
  k_bn_finalize<<<1, H_, 0, stream>>>(sum2, sq2, g2, be2, scale2, shift2);
  k_bn_apply<<<(B_ * H_ * N_) / 256, 256, 0, stream>>>(out, scale2, shift2);
}
